// BiSVM_89696097009955
// MI455X (gfx1250) — matrix operand order repaired and checked by reading, not compiled, not run on hardware
//
#include <hip/hip_runtime.h>

// out[b,o] = sum_{i,j} L[o,i] * x[b,i,j] * R[j,o]
// Pass 1 (if ws fits): split x (f32) into bf16 hi/lo planes in d_ws.
// Pass 2: per (b, o-tile16) WG, D = x[b] @ R[:,o-tile] via split-bf16 WMMA
// (hh + hl + lh, f32 accumulate), then out = sum_i L[o,i]*D[i,o] in exact f32.

typedef __attribute__((ext_vector_type(16))) __bf16 v16bf;
typedef __attribute__((ext_vector_type(8)))  __bf16 v8bf;
typedef __attribute__((ext_vector_type(4)))  __bf16 v4bf;
typedef __attribute__((ext_vector_type(8)))  float  v8f;

#define TILE_O  16
#define WAVES   8
#define THREADS 256

// ---------------- Pass 1: x -> (hi, lo) bf16 planes ----------------
__global__ __launch_bounds__(256)
void split_x_kernel(const float* __restrict__ X, __bf16* __restrict__ H,
                    __bf16* __restrict__ Lo, long n4) {
  long stride = (long)gridDim.x * blockDim.x;
  for (long i = (long)blockIdx.x * blockDim.x + threadIdx.x; i < n4; i += stride) {
    float4 v = ((const float4*)X)[i];
    float f[4] = {v.x, v.y, v.z, v.w};
    v4bf h, l;
#pragma unroll
    for (int t = 0; t < 4; ++t) {
      __bf16 hh = (__bf16)f[t];
      h[t] = hh;
      l[t] = (__bf16)(f[t] - (float)hh);
    }
    ((v4bf*)H)[i]  = h;
    ((v4bf*)Lo)[i] = l;
  }
}

// ---------------- Pass 2: fused GEMM + diagonal contraction ----------------
__global__ __launch_bounds__(THREADS)
void BiSVM_main_kernel(const __bf16* __restrict__ XH, const __bf16* __restrict__ XL,
                       const float* __restrict__ L, const float* __restrict__ R,
                       float* __restrict__ out) {
  // 32 KB hi-R fragments + 32 KB lo-R fragments; tail-reused for reduction.
  __shared__ __align__(32) unsigned char lds[65536];

  const int o0 = blockIdx.x * TILE_O;
  const int b  = blockIdx.y;

  // Stage R[:, o0:o0+16] as bf16 hi/lo, pre-swizzled into the CDNA5 16-bit
// B-matrix (32x16) fragment layout: lane = n + 16*((k>>3)&1), slot s = (k&7) + 8*(k>>4).
  for (int e = threadIdx.x; e < 1024 * TILE_O; e += THREADS) {
    const int   j = e >> 4;
    const int   n = e & 15;
    const float v = R[((size_t)j << 10) + o0 + n];
    const __bf16 h = (__bf16)v;
    const __bf16 l = (__bf16)(v - (float)h);
    const int   k   = j & 31;
    const int   off = ((j >> 5) << 10) + ((n + (((k >> 3) & 1) << 4)) << 5) + (((k & 7) + ((k >> 4) << 3)) << 1);
    *(__bf16*)(lds + off)         = h;
    *(__bf16*)(lds + 32768 + off) = l;
  }
  __syncthreads();

  const int wave = threadIdx.x >> 5;
  const int lane = threadIdx.x & 31;
  const int n    = lane & 15;
  const int half = lane >> 4;

  float outacc = 0.f;

  for (int itile = wave; itile < 64; itile += WAVES) {
    const int i0 = itile << 4;
    const size_t rowoff = ((size_t)b << 20) + ((size_t)(i0 + n) << 10) + (half << 3);
    const __bf16* xh = XH + rowoff;
    const __bf16* xl = XL + rowoff;
    const float*  Lrow = L + ((size_t)(o0 + n) << 10) + i0 + (half << 3);
    __builtin_prefetch(Lrow, 0, 0);  // global_prefetch_b8

    v8f acc0 = {};  // hi*hi
    v8f acc1 = {};  // hi*lo + lo*hi
#pragma unroll 2
    for (int js = 0; js < 32; ++js) {
      const int j0 = js << 5;
      // A 16x32 bf16 fragment: slots 0..7 = K 8h..8h+7, 8..15 = K 16+8h..+7
      const v8bf ah0 = *(const v8bf*)(xh + j0);
      const v8bf ah1 = *(const v8bf*)(xh + j0 + 16);
      const v8bf al0 = *(const v8bf*)(xl + j0);
      const v8bf al1 = *(const v8bf*)(xl + j0 + 16);
      const v16bf ah = __builtin_shufflevector(ah0, ah1, 0,1,2,3,4,5,6,7,
                                               8,9,10,11,12,13,14,15);
      const v16bf al = __builtin_shufflevector(al0, al1, 0,1,2,3,4,5,6,7,
                                               8,9,10,11,12,13,14,15);

      const v16bf bh = *(const v16bf*)(lds + (js << 10) + (lane << 5));
      const v16bf bl = *(const v16bf*)(lds + 32768 + (js << 10) + (lane << 5));

      acc0 = __builtin_amdgcn_wmma_f32_16x16x32_bf16(false, ah, false, bh,
                                                     (short)0, acc0, false, false);
      acc1 = __builtin_amdgcn_wmma_f32_16x16x32_bf16(false, ah, false, bl,
                                                     (short)0, acc1, false, false);
      acc1 = __builtin_amdgcn_wmma_f32_16x16x32_bf16(false, al, false, bh,
                                                     (short)0, acc1, false, false);
    }

    // C/D layout: VGPR r -> (M = r + 8*half, N = n). Apply L[o,i] in f32.
    const float4 l0 = *(const float4*)(Lrow);
    const float4 l1 = *(const float4*)(Lrow + 4);
    outacc += l0.x * (acc0[0] + acc1[0]) + l0.y * (acc0[1] + acc1[1])
            + l0.z * (acc0[2] + acc1[2]) + l0.w * (acc0[3] + acc1[3])
            + l1.x * (acc0[4] + acc1[4]) + l1.y * (acc0[5] + acc1[5])
            + l1.z * (acc0[6] + acc1[6]) + l1.w * (acc0[7] + acc1[7]);
  }

  outacc += __shfl_xor(outacc, 16, 32);

  __syncthreads();
  float* red = (float*)lds;
  if (lane < 16) red[wave * 16 + n] = outacc;
  __syncthreads();
  if (threadIdx.x < 16) {
    float s = 0.f;
#pragma unroll
    for (int w = 0; w < WAVES; ++w) s += red[w * 16 + threadIdx.x];
    out[((size_t)b << 10) + o0 + threadIdx.x] = s;
  }
}

// ---------------- Fallback (no workspace): in-loop split ----------------
__device__ __forceinline__ void split8(const float4 p, const float4 q,
                                       v16bf& hi, v16bf& lo, const int base) {
  float f[8] = {p.x, p.y, p.z, p.w, q.x, q.y, q.z, q.w};
#pragma unroll
  for (int t = 0; t < 8; ++t) {
    __bf16 h = (__bf16)f[t];
    float  r = f[t] - (float)h;
    hi[base + t] = h;
    lo[base + t] = (__bf16)r;
  }
}

__global__ __launch_bounds__(THREADS)
void BiSVM_fallback_kernel(const float* __restrict__ X, const float* __restrict__ L,
                           const float* __restrict__ R, float* __restrict__ out) {
  __shared__ __align__(32) unsigned char lds[65536];
  const int o0 = blockIdx.x * TILE_O;
  const int b  = blockIdx.y;

  for (int e = threadIdx.x; e < 1024 * TILE_O; e += THREADS) {
    const int   j = e >> 4;
    const int   n = e & 15;
    const float v = R[((size_t)j << 10) + o0 + n];
    const __bf16 h = (__bf16)v;
    const __bf16 l = (__bf16)(v - (float)h);
    const int   k   = j & 31;
    const int   off = ((j >> 5) << 10) + ((n + (((k >> 3) & 1) << 4)) << 5) + (((k & 7) + ((k >> 4) << 3)) << 1);
    *(__bf16*)(lds + off)         = h;
    *(__bf16*)(lds + 32768 + off) = l;
  }
  __syncthreads();

  const int wave = threadIdx.x >> 5;
  const int lane = threadIdx.x & 31;
  const int n    = lane & 15;
  const int half = lane >> 4;
  float outacc = 0.f;

  for (int itile = wave; itile < 64; itile += WAVES) {
    const int i0 = itile << 4;
    const float* xrow = X + ((size_t)b << 20) + ((size_t)(i0 + n) << 10);
    const float* Lrow = L + ((size_t)(o0 + n) << 10) + i0 + (half << 3);
    v8f acc = {};
#pragma unroll 2
    for (int js = 0; js < 32; ++js) {
      const int j0 = js << 5;
      const float4 a0 = *(const float4*)(xrow + j0 + (half << 3));
      const float4 a1 = *(const float4*)(xrow + j0 + (half << 3) + 4);
      const float4 a2 = *(const float4*)(xrow + j0 + 16 + (half << 3));
      const float4 a3 = *(const float4*)(xrow + j0 + 16 + (half << 3) + 4);
      v16bf ah, al;
      split8(a0, a1, ah, al, 0);
      split8(a2, a3, ah, al, 8);
      const v16bf bh = *(const v16bf*)(lds + (js << 10) + (lane << 5));
      const v16bf bl = *(const v16bf*)(lds + 32768 + (js << 10) + (lane << 5));
      acc = __builtin_amdgcn_wmma_f32_16x16x32_bf16(false, ah, false, bh,
                                                    (short)0, acc, false, false);
      acc = __builtin_amdgcn_wmma_f32_16x16x32_bf16(false, ah, false, bl,
                                                    (short)0, acc, false, false);
      acc = __builtin_amdgcn_wmma_f32_16x16x32_bf16(false, al, false, bh,
                                                    (short)0, acc, false, false);
    }
    const float4 l0 = *(const float4*)(Lrow);
    const float4 l1 = *(const float4*)(Lrow + 4);
    outacc += l0.x * acc[0] + l0.y * acc[1] + l0.z * acc[2] + l0.w * acc[3]
            + l1.x * acc[4] + l1.y * acc[5] + l1.z * acc[6] + l1.w * acc[7];
  }

  outacc += __shfl_xor(outacc, 16, 32);
  __syncthreads();
  float* red = (float*)lds;
  if (lane < 16) red[wave * 16 + n] = outacc;
  __syncthreads();
  if (threadIdx.x < 16) {
    float s = 0.f;
#pragma unroll
    for (int w = 0; w < WAVES; ++w) s += red[w * 16 + threadIdx.x];
    out[((size_t)b << 10) + o0 + threadIdx.x] = s;
  }
}

extern "C" void kernel_launch(void* const* d_in, const int* in_sizes, int n_in,
                              void* d_out, int out_size, void* d_ws, size_t ws_size,
                              hipStream_t stream) {
  const float* X = (const float*)d_in[0];   // [B, 1024, 1024] f32
  const float* L = (const float*)d_in[1];   // [1024, 1024] f32
  const float* R = (const float*)d_in[2];   // [1024, 1024] f32
  float* out = (float*)d_out;               // [B, 1024] f32

  const long   N = (long)in_sizes[0];       // B * 1024 * 1024
  const int    B = (int)(N >> 20);
  const size_t need = (size_t)N * 4;        // two bf16 planes of x
  dim3 grid(1024 / TILE_O, B);
  dim3 block(THREADS);

  if (ws_size >= need) {
    __bf16* XH = (__bf16*)d_ws;
    __bf16* XL = XH + N;
    split_x_kernel<<<4096, 256, 0, stream>>>(X, XH, XL, N >> 2);
    BiSVM_main_kernel<<<grid, block, 0, stream>>>(XH, XL, L, R, out);
  } else {
    BiSVM_fallback_kernel<<<grid, block, 0, stream>>>(X, L, R, out);
  }
}
